// RestoreNet_rotate_back_90228672954715
// MI455X (gfx1250) — matrix operand order repaired and checked by reading, not compiled, not run on hardware
//
#include <hip/hip_runtime.h>
#include <hip/hip_bf16.h>

#define BB 8
#define NN 2048
#define DD 512
#define KK 16
#define EPSF 1e-8f

typedef __attribute__((ext_vector_type(16))) _Float16 v16h;
typedef __attribute__((ext_vector_type(8)))  _Float16 v8h;
typedef __attribute__((ext_vector_type(8)))  float    v8f;

// ---------------------------------------------------------------------------
// Kernel 1: L2-normalize ffq/ffp rows, pack to f16 into workspace.
// grid = B*N blocks, 128 threads. Each thread handles 4 dims.
// ---------------------------------------------------------------------------
__global__ void rn_normalize_f16(const float* __restrict__ ffq,
                                 const float* __restrict__ ffp,
                                 _Float16* __restrict__ fqh,
                                 _Float16* __restrict__ fph) {
    const int row = blockIdx.x;             // b*N + n
    const int t   = threadIdx.x;            // 0..127
    const size_t base = (size_t)row * DD;

    __shared__ float sq[128];
    __shared__ float sp[128];

    float aq = 0.f, ap = 0.f;
    #pragma unroll
    for (int j = 0; j < 4; ++j) {
        float vq = ffq[base + t + j * 128];
        float vp = ffp[base + t + j * 128];
        aq += vq * vq;
        ap += vp * vp;
    }
    sq[t] = aq; sp[t] = ap;
    __syncthreads();
    for (int s = 64; s > 0; s >>= 1) {
        if (t < s) { sq[t] += sq[t + s]; sp[t] += sp[t + s]; }
        __syncthreads();
    }
    const float scq = 1.0f / (sqrtf(sq[0]) + EPSF);
    const float scp = 1.0f / (sqrtf(sp[0]) + EPSF);

    #pragma unroll
    for (int j = 0; j < 4; ++j) {
        fqh[base + t + j * 128] = (_Float16)(ffq[base + t + j * 128] * scq);
        fph[base + t + j * 128] = (_Float16)(ffp[base + t + j * 128] * scp);
    }
}

// ---------------------------------------------------------------------------
// Kernel 2: similarity via WMMA + fused top-16 per query row.
// grid = B * (N/16) blocks, 128 threads (4 waves).
//
// Per block: 16 query rows. A fragments (16 K-chunks) preloaded to registers.
// Per iteration, wave w computes the 16x16 score tile for columns
// [iter*64 + w*16, +16): all 32 B-fragment b128 loads are batch-issued into
// a register array, then the 16-deep v_wmma_f32_16x16x32_f16 chain runs with
// only partial loadcnt waits (loads return in order -> software pipelining).
// The next tile's B rows are prefetched (global_prefetch_b8) before the
// barrier + top-k selection phase so that VALU work hides the L2 latency.
// ---------------------------------------------------------------------------
__global__ void __launch_bounds__(128)
rn_simtopk(const _Float16* __restrict__ fqh,
           const _Float16* __restrict__ fph,
           int* __restrict__ topidx) {
    const int tilesPerB = NN / 16;
    const int b     = blockIdx.x / tilesPerB;
    const int mbase = (blockIdx.x % tilesPerB) * 16;

    const int t    = threadIdx.x;
    const int lane = t & 31;
    const int wave = t >> 5;
    const int half = lane >> 4;      // 0: lanes 0-15, 1: lanes 16-31
    const int l16  = lane & 15;

    __shared__ float scores[16][64];     // [query][col within 64-col chunk]
    __shared__ float cand_v[16][128];    // merge buffers
    __shared__ int   cand_i[16][128];

    // ---- preload all 16 A fragments (query m = mbase + l16) ----
    const _Float16* aptr = fqh + ((size_t)(b * NN + mbase + l16)) * DD;
    const int kbA = half ? 16 : 0;
    v16h afr[16];
    #pragma unroll
    for (int dk = 0; dk < DD / 32; ++dk) {
        v8h a0 = *(const v8h*)(aptr + dk * 32 + kbA);
        v8h a1 = *(const v8h*)(aptr + dk * 32 + kbA + 8);
        afr[dk] = __builtin_shufflevector(a0, a1,
            0, 1, 2, 3, 4, 5, 6, 7, 8, 9, 10, 11, 12, 13, 14, 15);
    }

    // per-thread top-16 (register resident: all index math unrolled-constant)
    float bestv[16];
    int   besti[16];
    #pragma unroll
    for (int j = 0; j < 16; ++j) { bestv[j] = -1e30f; besti[j] = 0x7fffffff; }

    const int subq = t & 15;   // which query this thread selects for
    const int sub  = t >> 4;   // which 8-column slice (0..7)

    for (int iter = 0; iter < NN / 64; ++iter) {
        const int colbase = iter * 64 + wave * 16;
        const _Float16* brow =
            fph + ((size_t)(b * NN + colbase + l16)) * DD;
        const _Float16* bptr = brow + (half ? 16 : 0);

        // ---- batch-issue all B-fragment loads for this tile ----
        v16h bfr[16];
        #pragma unroll
        for (int dk = 0; dk < DD / 32; ++dk)
            bfr[dk] = *(const v16h*)(bptr + dk * 32);

        // ---- prefetch next column tile's B rows (8 cachelines/row across
        //      the two half-lane groups) ----
        if (iter + 1 < NN / 64) {
            const _Float16* nrow =
                fph + ((size_t)(b * NN + colbase + 64 + l16)) * DD + half * 64;
            #pragma unroll
            for (int j = 0; j < 4; ++j)
                __builtin_prefetch((const void*)(nrow + j * 128), 0, 1);
        }

        // ---- 16-deep WMMA chain (overlaps with in-flight B loads) ----
        v8f c = {};
        #pragma unroll
        for (int dk = 0; dk < DD / 32; ++dk)
            c = __builtin_amdgcn_wmma_f32_16x16x32_f16(
                    false, afr[dk], false, bfr[dk], (short)0, c, false, false);

        // C layout: VGPR v, lane -> (M = v + 8*half, Ncol = l16)
        #pragma unroll
        for (int v = 0; v < 8; ++v)
            scores[v + 8 * half][wave * 16 + l16] = c[v];
        __syncthreads();

        // selection: thread covers cols [sub*8, sub*8+8) of this 64-col chunk
        #pragma unroll
        for (int j = 0; j < 8; ++j) {
            const int col = sub * 8 + j;
            float sv = scores[subq][col];
            int   si = iter * 64 + col;
            if (sv > bestv[15]) {
                #pragma unroll
                for (int r = 0; r < 16; ++r) {
                    bool ins = sv > bestv[r];
                    float tv = bestv[r]; int ti = besti[r];
                    if (ins) { bestv[r] = sv; besti[r] = si; sv = tv; si = ti; }
                }
            }
        }
        __syncthreads();
    }

    // dump partial lists
    #pragma unroll
    for (int r = 0; r < 16; ++r) {
        cand_v[subq][sub * 16 + r] = bestv[r];
        cand_i[subq][sub * 16 + r] = besti[r];
    }
    __syncthreads();

    // merge: 16 threads, one per query; value desc, index asc on ties
    if (t < 16) {
        int* outp = topidx + ((size_t)(b * NN + mbase + t)) * KK;
        for (int r = 0; r < KK; ++r) {
            float mv = -1e30f; int mi = 0x7fffffff; int mslot = -1;
            for (int s = 0; s < 128; ++s) {
                float v = cand_v[t][s];
                int   i = cand_i[t][s];
                if (v > mv || (v == mv && i < mi)) { mv = v; mi = i; mslot = s; }
            }
            cand_v[t][mslot] = -1e38f;
            outp[r] = mi;
        }
    }
}

// ---------------------------------------------------------------------------
// Kernel 3: gather K neighbors, mean/max pool, refine matmul, 3x3 inverses,
// masked update, write all tuple outputs.
// grid = B*N blocks, 256 threads (each owns 2 feature dims).
// ---------------------------------------------------------------------------
__device__ __forceinline__ void inv3_apply(const float* R,
                                           float v0, float v1, float v2,
                                           float* o) {
    float a = R[0], b = R[1], c = R[2];
    float d = R[3], e = R[4], f = R[5];
    float g = R[6], h = R[7], i = R[8];
    float det = a * (e * i - f * h) - b * (d * i - f * g) + c * (d * h - e * g);
    float iv = 1.0f / det;
    float i00 = (e * i - f * h) * iv, i01 = (c * h - b * i) * iv, i02 = (b * f - c * e) * iv;
    float i10 = (f * g - d * i) * iv, i11 = (a * i - c * g) * iv, i12 = (c * d - a * f) * iv;
    float i20 = (d * h - e * g) * iv, i21 = (b * g - a * h) * iv, i22 = (a * e - b * d) * iv;
    o[0] = i00 * v0 + i01 * v1 + i02 * v2;
    o[1] = i10 * v0 + i11 * v1 + i12 * v2;
    o[2] = i20 * v0 + i21 * v1 + i22 * v2;
}

__global__ void __launch_bounds__(256)
rn_refine(const float* __restrict__ q,
          const float* __restrict__ ffp,
          const float* __restrict__ ffq,
          const float* __restrict__ R1,
          const float* __restrict__ R2,
          const int*   __restrict__ centroids,
          const float* __restrict__ W,
          const float* __restrict__ bias,
          const int*   __restrict__ topidx,
          float* __restrict__ out) {
    const int bn = blockIdx.x;          // b*N + n
    const int b  = bn / NN;
    const int t  = threadIdx.x;

    __shared__ int   nb[KK];
    __shared__ float red[3][256];

    if (t < KK) nb[t] = topidx[(size_t)bn * KK + t];
    __syncthreads();

    float acc0 = 0.f, acc1 = 0.f, acc2 = 0.f;
    #pragma unroll
    for (int j = 0; j < 2; ++j) {
        const int d = t * 2 + j;
        const float fqv = ffq[(size_t)bn * DD + d];
        float sum = 0.f, mx = -1e30f;
        #pragma unroll
        for (int k = 0; k < KK; ++k) {
            float v = ffp[((size_t)(b * NN + nb[k])) * DD + d];
            sum += v;
            mx = fmaxf(mx, v);
        }
        const float avg = sum * (1.0f / KK);
        #pragma unroll
        for (int o = 0; o < 3; ++o) {
            const float* wr = W + (size_t)o * (3 * DD);
            float cacc = wr[d] * fqv + wr[DD + d] * avg + wr[2 * DD + d] * mx;
            if (o == 0) acc0 += cacc;
            else if (o == 1) acc1 += cacc;
            else acc2 += cacc;
        }
    }
    red[0][t] = acc0; red[1][t] = acc1; red[2][t] = acc2;
    __syncthreads();
    for (int s = 128; s > 0; s >>= 1) {
        if (t < s) {
            red[0][t] += red[0][t + s];
            red[1][t] += red[1][t + s];
            red[2][t] += red[2][t + s];
        }
        __syncthreads();
    }

    const size_t o_q   = 0;
    const size_t o_r1o = (size_t)BB * NN * 3;
    const size_t o_r2o = (size_t)BB * NN * 3 * 2;
    const size_t o_idx = (size_t)BB * NN * 3 * 3;

    if (t == 0) {
        float v0 = red[0][0] + bias[0];
        float v1 = red[1][0] + bias[1];
        float v2 = red[2][0] + bias[2];
        float t1[3], t2[3];
        inv3_apply(R1 + (size_t)bn * 9, v0, v1, v2, t1);
        inv3_apply(R2 + (size_t)bn * 9, t1[0], t1[1], t1[2], t2);
        const float mask = (centroids[bn] >= NN) ? 1.0f : 0.0f;
        #pragma unroll
        for (int j = 0; j < 3; ++j) {
            const float qv = q[(size_t)bn * 3 + j];
            const float qr = qv + t2[j] * mask;
            out[o_q   + (size_t)bn * 3 + j] = qv;   // q passthrough
            out[o_r1o + (size_t)bn * 3 + j] = qr;   // q_refine
            out[o_r2o + (size_t)bn * 3 + j] = qr;   // q_refine (again)
        }
    }
    if (t < KK) {
        out[o_idx + (size_t)bn * KK + t] = (float)nb[t];
    }
}

// ---------------------------------------------------------------------------
extern "C" void kernel_launch(void* const* d_in, const int* in_sizes, int n_in,
                              void* d_out, int out_size, void* d_ws, size_t ws_size,
                              hipStream_t stream) {
    (void)in_sizes; (void)n_in; (void)out_size; (void)ws_size;

    const float* q_in      = (const float*)d_in[1];
    const float* ffp       = (const float*)d_in[2];
    const float* ffq       = (const float*)d_in[3];
    const float* R1        = (const float*)d_in[4];
    const float* R2        = (const float*)d_in[5];
    const int*   centroids = (const int*)d_in[6];
    const float* W         = (const float*)d_in[7];
    const float* bias      = (const float*)d_in[8];
    float*       out       = (float*)d_out;

    const size_t halfBytes = (size_t)BB * NN * DD * sizeof(_Float16); // 16 MiB
    char* ws = (char*)d_ws;
    _Float16* fqh    = (_Float16*)(ws);
    _Float16* fph    = (_Float16*)(ws + halfBytes);
    int*      topidx = (int*)(ws + 2 * halfBytes);

    // 1) normalize + pack f16
    rn_normalize_f16<<<BB * NN, 128, 0, stream>>>(ffq, ffp, fqh, fph);

    // 2) WMMA similarity + fused top-16
    rn_simtopk<<<BB * (NN / 16), 128, 0, stream>>>(fqh, fph, topidx);

    // 3) gather/pool/refine/inverse + outputs
    rn_refine<<<BB * NN, 256, 0, stream>>>(q_in, ffp, ffq, R1, R2, centroids,
                                           W, bias, topidx, out);
}
